// LoRAMultiHeadSelfAttention_30666066493853
// MI455X (gfx1250) — matrix operand order repaired and checked by reading, not compiled, not run on hardware
//
#include <hip/hip_runtime.h>

// ---------------------------------------------------------------------------
// LoRA MHSA forward for MI455X (gfx1250), bf16 WMMA path.
//   qkv  = x @ (W_qkv + 2 A_qkv B_qkv) + b_qkv          (bf16 WMMA GEMM)
//   attn = softmax(q k^T / 8) v  per (b,h)              (flash-style, WMMA)
//   out  = attn @ (W_proj + 2 A_proj B_proj) + b_proj   (bf16 WMMA GEMM, f32 out)
// ---------------------------------------------------------------------------

typedef __bf16 bf16_t;
typedef __attribute__((ext_vector_type(16))) __bf16 v16bf;
typedef __attribute__((ext_vector_type(8)))  __bf16 v8bf;
typedef __attribute__((ext_vector_type(8)))  float  v8f;
typedef __attribute__((ext_vector_type(4)))  float  v4f;
typedef __attribute__((ext_vector_type(4)))  int    v4i_t;

#define EMBED 1024
#define SEQ   2048
#define BATCH 4
#define NHEAD 16
#define DH    64
#define MROWS (BATCH * SEQ)   // 8192
#define NQKV  (3 * EMBED)     // 3072

#if __has_builtin(__builtin_amdgcn_global_load_async_to_lds_b128) && \
    __has_builtin(__builtin_amdgcn_s_wait_asynccnt)
#define HAVE_ASYNC_LDS 1
// builtin signature (probed): (AS1 v4i* src, AS3 v4i* dst, imm offset, imm cpol);
// the immediate offset applies to BOTH the global and the LDS address.
#define AS1P(p) ((__attribute__((address_space(1))) v4i_t*)(void*)(p))
#define AS3P(p) ((__attribute__((address_space(3))) v4i_t*)(void*)(p))
#else
#define HAVE_ASYNC_LDS 0
#endif

__device__ __forceinline__ v8f wmma_bf16(v16bf a, v16bf b, v8f c) {
  return __builtin_amdgcn_wmma_f32_16x16x32_bf16(false, a, false, b, (short)0, c,
                                                 false, false);
}
__device__ __forceinline__ v16bf cat8(v8bf lo, v8bf hi) {
  return __builtin_shufflevector(lo, hi, 0, 1, 2, 3, 4, 5, 6, 7,
                                 8, 9, 10, 11, 12, 13, 14, 15);
}

// --------------------------- fp32 -> bf16 convert ---------------------------
__global__ __launch_bounds__(256) void k_cvt_bf16(const float* __restrict__ in,
                                                  bf16_t* __restrict__ out, int n) {
  int i = (blockIdx.x * 256 + threadIdx.x) * 8;
  if (i + 7 < n) {
    v4f a = *(const v4f*)(in + i);
    v4f b = *(const v4f*)(in + i + 4);
    v8bf o;
    o[0] = (bf16_t)a.x; o[1] = (bf16_t)a.y; o[2] = (bf16_t)a.z; o[3] = (bf16_t)a.w;
    o[4] = (bf16_t)b.x; o[5] = (bf16_t)b.y; o[6] = (bf16_t)b.z; o[7] = (bf16_t)b.w;
    *(v8bf*)(out + i) = o;
  }
}

// ----------------- W_eff = W + 2*A@B  (rank-8), cast to bf16 ----------------
__global__ __launch_bounds__(256) void k_prep_weff(const float* __restrict__ W,
                                                   const float* __restrict__ A,
                                                   const float* __restrict__ Bm,
                                                   bf16_t* __restrict__ out,
                                                   int K, int N) {
  int idx = blockIdx.x * 256 + threadIdx.x;
  if (idx >= K * N) return;
  int k = idx / N, n = idx - k * N;
  float acc = W[idx];
#pragma unroll
  for (int r = 0; r < 8; ++r) acc += 2.0f * A[k * 8 + r] * Bm[r * N + n];
  out[idx] = (bf16_t)acc;
}

// --------------------------- tiled bf16 WMMA GEMM ---------------------------
// C[M,N] = A[M,K] (bf16, row-major) * Bw[K,N] (bf16, row-major) + bias[N]
// 256 threads = 8 waves; block tile 128x128, K-step 32; wave tile 64x32.
// A-tile: async global->LDS copy (ASYNCcnt).  B-tile: transposed via VGPRs.
// Epilogue staged through LDS in 4 passes of 32 rows -> coalesced b128 stores.
#define BM 128
#define BN 128
#define BK 32
#define LDA_P 40   // padded halfs per ldsA row (80B, 16B-aligned, conflict-free)
#define LDB_P 40   // padded halfs per ldsBt row
#define LDE_P 136  // padded f32 per ldsC row (544B)

template <bool OUT_BF>
__global__ __launch_bounds__(256) void k_gemm_bf16(
    const bf16_t* __restrict__ A, const bf16_t* __restrict__ Bw,
    const float* __restrict__ bias, void* __restrict__ outp,
    int M, int N, int K) {
  __shared__ bf16_t ldsA[BM * LDA_P];   // 10.0 KB
  __shared__ bf16_t ldsBt[BN * LDB_P];  // 10.0 KB, stored transposed [n][k]
  __shared__ float  ldsC[32 * LDE_P];   // 17.0 KB, epilogue staging

  const int tid = threadIdx.x;
  const int wave = tid >> 5, lane = tid & 31;
  const int lo = lane & 15, hi = lane >> 4;
  const int wm = wave >> 2, wn = wave & 3;  // 2 x 4 wave grid
  const int m0 = blockIdx.y * BM, n0 = blockIdx.x * BN;

  v8f acc[4][2] = {};

  // incremental global pointers for the tile loaders
  const int ar = tid >> 1, ac = (tid & 1) * 16;        // A: 16 halfs/thread
  const int bk = tid >> 3, bc = (tid & 7) * 16;        // B: 16 halfs/thread
  const bf16_t* ga = A + (size_t)(m0 + ar) * K + ac;
  const bf16_t* gb = Bw + (size_t)bk * N + n0 + bc;
  const size_t bstep = (size_t)BK * N;
  bf16_t* lda = &ldsA[ar * LDA_P + ac];

  for (int k0 = 0; k0 < K; k0 += BK) {
#if HAVE_ASYNC_LDS
    // A tile 128x32: async DMA, same +16B offset on both LDS and global side
    __builtin_amdgcn_global_load_async_to_lds_b128(AS1P(ga), AS3P(lda), 0, 0);
    __builtin_amdgcn_global_load_async_to_lds_b128(AS1P(ga), AS3P(lda), 16, 0);
#else
    {
      v8bf a0 = *(const v8bf*)ga;
      v8bf a1 = *(const v8bf*)(ga + 8);
      *(v8bf*)lda = a0;
      *(v8bf*)(lda + 8) = a1;
    }
#endif
    {  // B tile 32x128 -> LDS transposed [n][k]
      v8bf b0 = *(const v8bf*)gb;
      v8bf b1 = *(const v8bf*)(gb + 8);
#pragma unroll
      for (int i = 0; i < 8; ++i) ldsBt[(bc + i) * LDB_P + bk] = b0[i];
#pragma unroll
      for (int i = 0; i < 8; ++i) ldsBt[(bc + 8 + i) * LDB_P + bk] = b1[i];
    }
    ga += BK;
    gb += bstep;
    if (k0 + BK < K) {  // prefetch next tiles into cache (global_prefetch_b8)
      __builtin_prefetch(ga, 0, 3);
      __builtin_prefetch(gb, 0, 3);
    }
#if HAVE_ASYNC_LDS
    __builtin_amdgcn_s_wait_asynccnt(0);
#endif
    __syncthreads();

    v16bf bfrag[2];
#pragma unroll
    for (int tn = 0; tn < 2; ++tn) {  // B frag: lane=n, 16 contiguous k
      const bf16_t* p = &ldsBt[(wn * 32 + tn * 16 + lo) * LDB_P + hi * 16];
      bfrag[tn] = cat8(*(const v8bf*)p, *(const v8bf*)(p + 8));
    }
#pragma unroll
    for (int tm = 0; tm < 4; ++tm) {  // A frag: lane=m, 16 contiguous k
      const bf16_t* p = &ldsA[(wm * 64 + tm * 16 + lo) * LDA_P + hi * 16];
      v16bf afrag = cat8(*(const v8bf*)p, *(const v8bf*)(p + 8));
      acc[tm][0] = wmma_bf16(afrag, bfrag[0], acc[tm][0]);
      acc[tm][1] = wmma_bf16(afrag, bfrag[1], acc[tm][1]);
    }
    __syncthreads();
  }

  // ---- epilogue: 4 passes of 32 rows through LDS, coalesced stores ----
  float bv[2];
  bv[0] = bias[n0 + wn * 32 + lo];
  bv[1] = bias[n0 + wn * 32 + 16 + lo];

  const int er = tid >> 3, ec = (tid & 7) * 16;  // copy-out: 16 f32 / thread
  char* dst = (char*)outp +
              ((size_t)(m0 + er) * N + n0 + ec) * (OUT_BF ? 2 : 4);
  const size_t dstep = (size_t)32 * N * (OUT_BF ? 2 : 4);

#pragma unroll
  for (int p = 0; p < 4; ++p) {
    __syncthreads();
    if (wm == (p >> 1)) {  // this wave owns rows [p*32, p*32+32)
#pragma unroll
      for (int t = 0; t < 2; ++t) {
        int tm = (p & 1) * 2 + t;
#pragma unroll
        for (int tn = 0; tn < 2; ++tn) {
          int col = wn * 32 + tn * 16 + lo;
#pragma unroll
          for (int r = 0; r < 8; ++r)
            ldsC[(t * 16 + hi * 8 + r) * LDE_P + col] = acc[tm][tn][r] + bv[tn];
        }
      }
    }
    __syncthreads();
    const float* src = &ldsC[er * LDE_P + ec];
    if (OUT_BF) {
      v8bf o0, o1;
#pragma unroll
      for (int i = 0; i < 8; ++i) o0[i] = (bf16_t)src[i];
#pragma unroll
      for (int i = 0; i < 8; ++i) o1[i] = (bf16_t)src[8 + i];
      *(v8bf*)dst = o0;
      *(v8bf*)(dst + 16) = o1;
    } else {
      // final output: never re-read -> non-temporal stores
#pragma unroll
      for (int i = 0; i < 4; ++i)
        __builtin_nontemporal_store(*(const v4f*)(src + i * 4),
                                    (v4f*)(dst + i * 16));
    }
    dst += dstep;
  }
}

// ------------------------- flash attention (per b,h) ------------------------
// grid.x = B*H (64), grid.y = SEQ/128; 8 waves x 16 Q-rows; Bc = 64.
// Softmax scale 1/8 folded into Q fragments (power of two => exact in bf16).
// Row-sums of P computed by WMMA against an all-ones B matrix (no shuffles).
#define ATBR 128
#define ATBC 64
#define LDK_P 72  // 144B rows
#define LDV_P 72
#define LDP_P 72

__global__ __launch_bounds__(256) void k_attn(const bf16_t* __restrict__ qkv,
                                              bf16_t* __restrict__ outp) {
  __shared__ bf16_t ldsK[ATBC * LDK_P];        // 9 KB
  __shared__ bf16_t ldsVt[DH * LDV_P];         // 9 KB, transposed [d][n]
  __shared__ bf16_t ldsP[8 * 16 * LDP_P];      // 18 KB, per-wave P tiles

  const int bh = blockIdx.x;
  const int b = bh >> 4, h = bh & 15;
  const int tid = threadIdx.x;
  const int wave = tid >> 5, lane = tid & 31;
  const int lo = lane & 15, hi = lane >> 4;
  const int C3 = NQKV;

  const bf16_t* base = qkv + (size_t)(b * SEQ) * C3;
  const int qrow0 = blockIdx.y * ATBR + wave * 16;

  // Q fragments (A-layout), pre-scaled by 1/8 (exact bf16 exponent shift)
  v16bf qfrag[2];
#pragma unroll
  for (int s = 0; s < 2; ++s) {
    const bf16_t* p = base + (size_t)(qrow0 + lo) * C3 + h * 64 + s * 32 + hi * 16;
    v16bf q = cat8(*(const v8bf*)p, *(const v8bf*)(p + 8));
#pragma unroll
    for (int i = 0; i < 16; ++i) q[i] = (bf16_t)((float)q[i] * 0.125f);
    qfrag[s] = q;
  }

  v16bf ones;
#pragma unroll
  for (int i = 0; i < 16; ++i) ones[i] = (bf16_t)1.0f;

  v8f oacc[4] = {};
  float m_i[8], l_i[8];
#pragma unroll
  for (int r = 0; r < 8; ++r) { m_i[r] = -1e30f; l_i[r] = 0.f; }

  bf16_t* pp = &ldsP[wave * 16 * LDP_P];

  // incremental K/V tile pointers
  const int kr = tid >> 2, kc = (tid & 3) * 16;  // 16 halfs/thread
  const bf16_t* gk = base + (size_t)kr * C3 + EMBED + h * 64 + kc;
  const bf16_t* gv = base + (size_t)kr * C3 + 2 * EMBED + h * 64 + kc;
  const size_t jstep = (size_t)ATBC * C3;
  bf16_t* ldk = &ldsK[kr * LDK_P + kc];

  for (int j0 = 0; j0 < SEQ; j0 += ATBC) {
#if HAVE_ASYNC_LDS
    // K tile 64x64 -> ldsK[n][d]: async DMA (32B per thread)
    __builtin_amdgcn_global_load_async_to_lds_b128(AS1P(gk), AS3P(ldk), 0, 0);
    __builtin_amdgcn_global_load_async_to_lds_b128(AS1P(gk), AS3P(ldk), 16, 0);
#else
    {
      v8bf a0 = *(const v8bf*)gk;
      v8bf a1 = *(const v8bf*)(gk + 8);
      *(v8bf*)ldk = a0;
      *(v8bf*)(ldk + 8) = a1;
    }
#endif
    {  // V tile 64x64 -> ldsVt[d][n] (transpose on store)
      v8bf a0 = *(const v8bf*)gv;
      v8bf a1 = *(const v8bf*)(gv + 8);
#pragma unroll
      for (int i = 0; i < 8; ++i) ldsVt[(kc + i) * LDV_P + kr] = a0[i];
#pragma unroll
      for (int i = 0; i < 8; ++i) ldsVt[(kc + 8 + i) * LDV_P + kr] = a1[i];
    }
    gk += jstep;
    gv += jstep;
    if (j0 + ATBC < SEQ) {
      __builtin_prefetch(gk, 0, 3);
      __builtin_prefetch(gv, 0, 3);
    }
#if HAVE_ASYNC_LDS
    __builtin_amdgcn_s_wait_asynccnt(0);
#endif
    __syncthreads();

    // S = (Q/8) K^T   (4 n-subtiles of 16; scale already in Q)
    v8f sacc[4];
#pragma unroll
    for (int nt = 0; nt < 4; ++nt) {
      v8f s = {};
#pragma unroll
      for (int ks = 0; ks < 2; ++ks) {
        const bf16_t* p = &ldsK[(nt * 16 + lo) * LDK_P + ks * 32 + hi * 16];
        v16bf kf = cat8(*(const v8bf*)p, *(const v8bf*)(p + 8));
        s = wmma_bf16(qfrag[ks], kf, s);
      }
      sacc[nt] = s;
    }

    // row max across tiles + 16 lanes (row m = hi*8 + r lives in lanes hi*16..+15)
    float rmax[8];
#pragma unroll
    for (int r = 0; r < 8; ++r) {
      float v01 = fmaxf(sacc[0][r], sacc[1][r]);
      float v23 = fmaxf(sacc[2][r], sacc[3][r]);
      rmax[r] = fmaxf(v01, v23);
    }
#pragma unroll
    for (int msk = 1; msk < 16; msk <<= 1)
#pragma unroll
      for (int r = 0; r < 8; ++r)
        rmax[r] = fmaxf(rmax[r], __shfl_xor(rmax[r], msk, 32));

    // online softmax update
    float alpha[8];
#pragma unroll
    for (int r = 0; r < 8; ++r) {
      float mn = fmaxf(m_i[r], rmax[r]);
      alpha[r] = __expf(m_i[r] - mn);
      m_i[r] = mn;
    }
#pragma unroll
    for (int nt = 0; nt < 4; ++nt)
#pragma unroll
      for (int r = 0; r < 8; ++r) {
        float e = __expf(sacc[nt][r] - m_i[r]);
        pp[(hi * 8 + r) * LDP_P + nt * 16 + lo] = (bf16_t)e;  // P, row-major
      }
#pragma unroll
    for (int dt = 0; dt < 4; ++dt)
#pragma unroll
      for (int r = 0; r < 8; ++r) oacc[dt][r] *= alpha[r];

    // load P fragments (A-layout) once; per-wave region, no cross-wave sync
    v16bf pf[2];
#pragma unroll
    for (int s = 0; s < 2; ++s) {
      const bf16_t* p = &pp[lo * LDP_P + s * 32 + hi * 16];
      pf[s] = cat8(*(const v8bf*)p, *(const v8bf*)(p + 8));
    }

    // row-sum of P via WMMA vs all-ones B: D[m][*] = sum_k P[m][k]
    v8f rs = {};
    rs = wmma_bf16(pf[0], ones, rs);
    rs = wmma_bf16(pf[1], ones, rs);
#pragma unroll
    for (int r = 0; r < 8; ++r) l_i[r] = l_i[r] * alpha[r] + rs[r];

    // O += P x V
#pragma unroll
    for (int s = 0; s < 2; ++s) {
#pragma unroll
      for (int dt = 0; dt < 4; ++dt) {
        const bf16_t* vp = &ldsVt[(dt * 16 + lo) * LDV_P + s * 32 + hi * 16];
        v16bf vf = cat8(*(const v8bf*)vp, *(const v8bf*)(vp + 8));
        oacc[dt] = wmma_bf16(pf[s], vf, oacc[dt]);
      }
    }
    __syncthreads();
  }

  // normalize, stage bf16 O tile in per-wave LDS, then coalesced 64B stores.
  float inv[8];
#pragma unroll
  for (int r = 0; r < 8; ++r) inv[r] = 1.0f / l_i[r];
#pragma unroll
  for (int dt = 0; dt < 4; ++dt)
#pragma unroll
    for (int r = 0; r < 8; ++r)
      pp[(hi * 8 + r) * LDP_P + dt * 16 + lo] = (bf16_t)(oacc[dt][r] * inv[r]);
  {
    int row = lane >> 1, c = (lane & 1) * 32;
    const bf16_t* src = &pp[row * LDP_P + c];
    bf16_t* dst = outp + (size_t)(b * SEQ + qrow0 + row) * EMBED + h * 64 + c;
    v8bf t0 = *(const v8bf*)(src);
    v8bf t1 = *(const v8bf*)(src + 8);
    v8bf t2 = *(const v8bf*)(src + 16);
    v8bf t3 = *(const v8bf*)(src + 24);
    *(v8bf*)(dst) = t0;
    *(v8bf*)(dst + 8) = t1;
    *(v8bf*)(dst + 16) = t2;
    *(v8bf*)(dst + 24) = t3;
  }
}

// --------------------------------- launch -----------------------------------
extern "C" void kernel_launch(void* const* d_in, const int* in_sizes, int n_in,
                              void* d_out, int out_size, void* d_ws, size_t ws_size,
                              hipStream_t stream) {
  (void)in_sizes; (void)n_in; (void)out_size; (void)ws_size;
  const float* x      = (const float*)d_in[0];
  const float* W_qkv  = (const float*)d_in[1];
  const float* b_qkv  = (const float*)d_in[2];
  const float* A_qkv  = (const float*)d_in[3];
  const float* B_qkv  = (const float*)d_in[4];
  const float* W_proj = (const float*)d_in[5];
  const float* b_proj = (const float*)d_in[6];
  const float* A_proj = (const float*)d_in[7];
  const float* B_proj = (const float*)d_in[8];

  // workspace layout (all 256B aligned); total = 92,274,688 bytes
  char* w = (char*)d_ws;
  bf16_t* x_bf   = (bf16_t*)(w);                       // 16,777,216
  bf16_t* weff_q = (bf16_t*)(w + 16777216);            //  6,291,456
  bf16_t* weff_p = (bf16_t*)(w + 23068672);            //  2,097,152
  bf16_t* qkv    = (bf16_t*)(w + 25165824);            // 50,331,648
  bf16_t* attn   = (bf16_t*)(w + 75497472);            // 16,777,216

  // 1) convert x to bf16
  k_cvt_bf16<<<(MROWS * EMBED) / (256 * 8), 256, 0, stream>>>(x, x_bf,
                                                              MROWS * EMBED);
  // 2) fold LoRA into weights: W_eff = W + 2 A@B, cast bf16
  k_prep_weff<<<(EMBED * NQKV + 255) / 256, 256, 0, stream>>>(
      W_qkv, A_qkv, B_qkv, weff_q, EMBED, NQKV);
  k_prep_weff<<<(EMBED * EMBED + 255) / 256, 256, 0, stream>>>(
      W_proj, A_proj, B_proj, weff_p, EMBED, EMBED);
  // 3) qkv GEMM (bf16 out)
  k_gemm_bf16<true><<<dim3(NQKV / BN, MROWS / BM), 256, 0, stream>>>(
      x_bf, weff_q, b_qkv, (void*)qkv, MROWS, NQKV, EMBED);
  // 4) flash attention
  k_attn<<<dim3(BATCH * NHEAD, SEQ / ATBR), 256, 0, stream>>>(qkv, attn);
  // 5) projection GEMM (f32 out)
  k_gemm_bf16<false><<<dim3(EMBED / BN, MROWS / BM), 256, 0, stream>>>(
      attn, weff_p, b_proj, (void*)d_out, MROWS, EMBED, EMBED);
}
